// Attention_59974923321367
// MI455X (gfx1250) — matrix operand order repaired and checked by reading, not compiled, not run on hardware
//
// FAVOR+ causal linear attention for MI455X (gfx1250), chunked-state formulation.
// All matmuls via v_wmma_f32_16x16x32_bf16 (wave32, 16x16x32 tiles, f32 accum).
// K3c tile movement via the Tensor Data Mover (tensor_load_to_lds /
// tensor_store_from_lds + s_wait_tensorcnt) when the builtin is available.
//
// Pipeline:
//  K1 x3 : q/k/v = x @ W{q,k,v}                    (GEMM 16384x1024x1024, bf16 out)
//  K2 x2 : qp/kp = relu((1/16) q @ P^T) + 1e-3     (GEMM 262144x256x64, bf16 out)
//  K3a   : per-(b,h,chunk) local state kp^T @ [v|1|0] -> states (M=256 x Dext=80)
//  K3b   : per-(b,h) exclusive prefix scan over 64 chunk states, in-place,
//          transposed into B-operand layout [Dext][M]
//  K3c   : per-(b,h,chunk): Y = qp@state + causal(qp@kp^T)@[v|1|0]; out = num/den
//  K4    : out = attn @ Wo                          (GEMM 16384x1024x1024, f32 out)
//
// Workspace (bf16): q,k,v (3*16Mi) + qp,kp (2*64Mi) + attn (16Mi) + states (80Mi)
//  = 285,212,672 elems = ~544 MiB.

#include <hip/hip_runtime.h>

typedef __attribute__((ext_vector_type(16))) __bf16 v16bf;
typedef __attribute__((ext_vector_type(8)))  float  v8f;

constexpr int Bb   = 4;
constexpr int Ld   = 4096;
constexpr int Hh   = 16;
constexpr int Dd   = 64;
constexpr int Mm   = 256;     // feature count
constexpr int HD   = 1024;    // hidden = H*D
constexpr int CHK  = 64;      // chunk length
constexpr int NCHK = Ld / CHK;
constexpr int DEXT = 80;      // D + denominator column + pad to 5 tiles
constexpr int SLOTS = Mm * DEXT;   // 20480 per chunk state

// ---------------- Tensor Data Mover (CDNA5 TDM) support ---------------------
#if defined(__HIP_DEVICE_COMPILE__) && defined(__has_builtin)
#if __has_builtin(__builtin_amdgcn_tensor_load_to_lds) && \
    __has_builtin(__builtin_amdgcn_tensor_store_from_lds) && \
    __has_builtin(__builtin_amdgcn_s_wait_tensorcnt)
#define HAVE_TDM 1
#endif
#endif
#ifndef HAVE_TDM
#define HAVE_TDM 0
#endif

#if HAVE_TDM
typedef __attribute__((ext_vector_type(4))) unsigned int v4u_t;
typedef __attribute__((ext_vector_type(8))) int v8i_t;
typedef __attribute__((ext_vector_type(4))) int v4i_t;

// Build a 2D D# (ISA 8.3/8.4): bf16 elements (data_size=2B), tile = tensor =
// tw x th elements, row stride `stride` elements. lds_off = byte offset into
// the workgroup's dynamic LDS allocation (which starts at 0 here).
__device__ inline void tdm_desc(v4u_t& g0, v8i_t& g1, unsigned lds_off,
                                const void* gptr, unsigned tw, unsigned th,
                                unsigned stride) {
  unsigned long long ga = (unsigned long long)(size_t)gptr;
  g0[0] = 1u;                                            // count=1, user D#
  g0[1] = lds_off;                                       // lds_addr (bytes)
  g0[2] = (unsigned)ga;                                  // global_addr[31:0]
  g0[3] = (unsigned)((ga >> 32) & 0x01ffffffu) | (2u << 30); // addr[56:32]|type=2
  g1[0] = (int)(1u << 16);                               // data_size=1 -> 2 bytes
  g1[1] = (int)((tw & 0xffffu) << 16);                   // tensor_dim0 lo
  g1[2] = (int)((tw >> 16) | ((th & 0xffffu) << 16));    // tensor_dim0 hi|dim1 lo
  g1[3] = (int)((th >> 16) | ((tw & 0xffffu) << 16));    // dim1 hi | tile_dim0
  g1[4] = (int)(th & 0xffffu);                           // tile_dim1 (tile_dim2=0)
  g1[5] = (int)stride;                                   // tensor_dim0_stride lo
  g1[6] = 0;
  g1[7] = 0;
}

// clang-23 / therock-10.0 arity: (v4u, v8i, v4i, v4i, v8i, i32 cpol)
__device__ inline void tdm_load_2d(unsigned lds_off, const void* gptr,
                                   unsigned tw, unsigned th, unsigned stride) {
  v4u_t g0; v8i_t g1; v4i_t z4 = {}; v8i_t z8 = {};
  tdm_desc(g0, g1, lds_off, gptr, tw, th, stride);
  __builtin_amdgcn_tensor_load_to_lds(g0, g1, z4, z4, z8, 0);
}
__device__ inline void tdm_store_2d(unsigned lds_off, void* gptr,
                                    unsigned tw, unsigned th, unsigned stride) {
  v4u_t g0; v8i_t g1; v4i_t z4 = {}; v8i_t z8 = {};
  tdm_desc(g0, g1, lds_off, gptr, tw, th, stride);
  __builtin_amdgcn_tensor_store_from_lds(g0, g1, z4, z4, z8, 0);
}
#endif

__device__ inline v8f wmma_bf16(v16bf a, v16bf b, v8f c) {
  return __builtin_amdgcn_wmma_f32_16x16x32_bf16(false, a, false, b, (short)0, c,
                                                 false, false);
}

// A fragment, 16x32 bf16, from row-major [rows][ld] tile. ISA 7.12.2:
// lane m=lane&15, half h=lane>>4: elems 0..7 -> K=8h..8h+7, elems 8..15 -> K=16+8h..
__device__ inline v16bf lda_frag(const __bf16* base, int row0, int k0, int ld) {
  int lane = (int)(threadIdx.x & 31u);
  const __bf16* p = base + (size_t)(row0 + (lane & 15)) * ld + k0 + 8 * (lane >> 4);
  v16bf r;
#pragma unroll
  for (int e = 0; e < 8; ++e) { r[e] = p[e]; r[e + 8] = p[16 + e]; }
  return r;
}

// B fragment, 32x16 bf16, from B^T stored row-major [N][ld] (ld = K stride).
// lane n=lane&15, half h=lane>>4: elems 0..7 -> K=8h..8h+7, elems 8..15 -> K=16+8h..
__device__ inline v16bf ldb_frag(const __bf16* base, int n0, int k0, int ld) {
  int lane = (int)(threadIdx.x & 31u);
  const __bf16* p = base + (size_t)(n0 + (lane & 15)) * ld + k0 + 8 * (lane >> 4);
  v16bf r;
#pragma unroll
  for (int e = 0; e < 8; ++e) { r[e] = p[e]; r[e + 8] = p[16 + e]; }
  return r;
}

// ---------------------------------------------------------------------------
// Generic tiled GEMM: C[Mr x Nc] = epilogue(A[Mr x Kc] * B), bf16 WMMA compute.
// B is fp32; BT=false: B is [Kc][Nc] row-major; BT=true: B is [Nc][Kc] (pre-transposed).
// RELU epilogue: y = max(alpha*y, 0) + beta.
// Block: 256 thr = 8 waves, macro-tile 128x64, K-step 32, each wave 32x32.
// ---------------------------------------------------------------------------
template <typename TA, typename TC, bool BT, bool RELU>
__global__ void __launch_bounds__(256)
gemm_kernel(const TA* __restrict__ A, const float* __restrict__ B,
            TC* __restrict__ C, int Mr, int Nc, int Kc, float alpha, float beta) {
  __shared__ __bf16 As[128 * 32];
  __shared__ __bf16 Bs[64 * 32];
  const int t = (int)threadIdx.x;
  const int w = t >> 5, lane = t & 31, hl = lane >> 4, col = lane & 15;
  const int wr = w >> 1, wc = w & 1;
  const int m0 = (int)blockIdx.y * 128, n0 = (int)blockIdx.x * 64;

  v8f acc[2][2] = {};
  for (int kb = 0; kb < Kc; kb += 32) {
#pragma unroll
    for (int i = 0; i < 16; ++i) {
      int idx = i * 256 + t, ar = idx >> 5, ak = idx & 31;
      As[ar * 32 + ak] = (__bf16)(float)A[(size_t)(m0 + ar) * Kc + kb + ak];
    }
#pragma unroll
    for (int i = 0; i < 8; ++i) {
      int idx = i * 256 + t;
      if (BT) {
        int bn = idx >> 5, bk = idx & 31;
        Bs[bn * 32 + bk] = (__bf16)B[(size_t)(n0 + bn) * Kc + kb + bk];
      } else {
        int bk = idx >> 6, bn = idx & 63;
        Bs[bn * 32 + bk] = (__bf16)B[(size_t)(kb + bk) * Nc + n0 + bn];
      }
    }
    if (kb + 32 < Kc)  // hint next A slab into cache (global_prefetch_b8)
      __builtin_prefetch(A + (size_t)(m0 + (t >> 1)) * Kc + kb + 32, 0, 1);
    __syncthreads();
    v16bf a0 = lda_frag(As, wr * 32, 0, 32);
    v16bf a1 = lda_frag(As, wr * 32 + 16, 0, 32);
    v16bf b0 = ldb_frag(Bs, wc * 32, 0, 32);
    v16bf b1 = ldb_frag(Bs, wc * 32 + 16, 0, 32);
    acc[0][0] = wmma_bf16(a0, b0, acc[0][0]);
    acc[0][1] = wmma_bf16(a0, b1, acc[0][1]);
    acc[1][0] = wmma_bf16(a1, b0, acc[1][0]);
    acc[1][1] = wmma_bf16(a1, b1, acc[1][1]);
    __syncthreads();
  }
#pragma unroll
  for (int i = 0; i < 2; ++i) {
#pragma unroll
    for (int j = 0; j < 2; ++j) {
#pragma unroll
      for (int r = 0; r < 8; ++r) {
        int row = m0 + wr * 32 + i * 16 + r + 8 * hl;
        int cg  = n0 + wc * 32 + j * 16 + col;
        float y = acc[i][j][r];
        if (RELU) y = fmaxf(y * alpha, 0.f) + beta;
        C[(size_t)row * Nc + cg] = (TC)y;
      }
    }
  }
}

// ---------------------------------------------------------------------------
// K3a: local chunk state  S_local[b,h,c][m][dext] = sum_l kp[l][m]*v_ext[l][dext]
// v_ext: cols 0..63 = v, col 64 = 1.0 (denominator), 65..79 = 0.
// grid = B*H*NCHK, block = 256 (8 waves, 80 tiles -> 10 per wave).
// ---------------------------------------------------------------------------
__global__ void __launch_bounds__(256)
local_state_kernel(const __bf16* __restrict__ kp, const __bf16* __restrict__ v,
                   __bf16* __restrict__ states) {
  extern __shared__ char smem[];
  __bf16* kpT = (__bf16*)smem;          // [M=256][C=64]   (A operand layout)
  __bf16* vT  = kpT + 256 * 64;         // [DEXT=80][C=64] (B^T operand layout)
  const int t = (int)threadIdx.x, lane = t & 31, w = t >> 5;
  const int hl = lane >> 4, col = lane & 15;
  const int bid = (int)blockIdx.x;
  const int c = bid & (NCHK - 1), h = (bid >> 6) & (Hh - 1), b = bid >> 10;
  const int lbase = b * Ld + c * CHK;

  for (int i = 0; i < 64; ++i) {
    int idx = i * 256 + t, l = idx >> 8, m = idx & 255;
    kpT[m * 64 + l] = kp[((size_t)((lbase + l) * Hh + h)) * Mm + m];
  }
  for (int i = 0; i < 16; ++i) {
    int idx = i * 256 + t, l = idx >> 6, d = idx & 63;
    vT[d * 64 + l] = v[(size_t)(lbase + l) * HD + h * Dd + d];
  }
  for (int i = 0; i < 4; ++i) {
    int idx = i * 256 + t, rr = 64 + (idx >> 6);
    vT[rr * 64 + (idx & 63)] = (__bf16)((rr == 64) ? 1.0f : 0.0f);
  }
  __syncthreads();

  v8f acc[10] = {};
  for (int kk = 0; kk < CHK; kk += 32) {
#pragma unroll
    for (int i = 0; i < 10; ++i) {
      int tt = w + 8 * i, tm = tt / 5, tn = tt % 5;
      v16bf a  = lda_frag(kpT, tm * 16, kk, 64);
      v16bf bb = ldb_frag(vT, tn * 16, kk, 64);
      acc[i] = wmma_bf16(a, bb, acc[i]);
    }
  }
  size_t gbase = (size_t)bid * SLOTS;
#pragma unroll
  for (int i = 0; i < 10; ++i) {
    int tt = w + 8 * i, tm = tt / 5, tn = tt % 5;
#pragma unroll
    for (int r = 0; r < 8; ++r) {
      int m = tm * 16 + r + 8 * hl, d = tn * 16 + col;
      states[gbase + (size_t)m * DEXT + d] = (__bf16)acc[i][r];
    }
  }
}

// ---------------------------------------------------------------------------
// K3b: per-(b,h) in-place EXCLUSIVE prefix scan over NCHK chunk states, writing
// result transposed to [dext][m] (the B^T layout K3c consumes). LDS-staged so
// global reads and writes both stay coalesced. grid = B*H, block = 256.
// ---------------------------------------------------------------------------
__global__ void __launch_bounds__(256)
scan_states_kernel(__bf16* __restrict__ states) {
  extern __shared__ char smem[];
  float* xch = (float*)smem;            // SLOTS floats (80 KB)
  const int t = (int)threadIdx.x;
  size_t base = (size_t)blockIdx.x * NCHK * SLOTS;
  float acc[80];
#pragma unroll
  for (int i = 0; i < 80; ++i) acc[i] = 0.f;
  for (int c = 0; c < NCHK; ++c) {
    size_t cb = base + (size_t)c * SLOTS;
    float tmp[80];
    for (int i = 0; i < 80; ++i) tmp[i] = (float)states[cb + i * 256 + t]; // s = m*80+d
    for (int i = 0; i < 80; ++i) xch[i * 256 + t] = acc[i];
    __syncthreads();
    for (int i = 0; i < 80; ++i) {
      int ws = i * 256 + t;             // write slot = d*256 + m
      int dd = ws >> 8, m = ws & 255;
      states[cb + ws] = (__bf16)xch[m * DEXT + dd];
    }
    __syncthreads();
    for (int i = 0; i < 80; ++i) acc[i] += tmp[i];
  }
}

// ---------------------------------------------------------------------------
// K3c: per-(b,h,chunk) output.
//   Y[C][DEXT]  = qp @ state                      (inter-chunk, K=256)
//               + causalmask(qp@kp^T) @ v_ext     (intra-chunk)
//   out[l][d]   = Y[l][d] / Y[l][64]
// grid = B*H*NCHK, block = 256. qp/kp/state tiles arrive via TDM.
// LDS layout (byte offsets within dynamic allocation, base assumed 0):
//   qpL 0      [64][256] bf16   (32768 B)
//   kpL 32768  [64][256] bf16   (32768 B)
//   PsL 65536  [80][256] bf16   (40960 B)
//   vT  106496 [80][64]  bf16   (10240 B)
//   Sm  116736 [64][64]  bf16   ( 8192 B)  scores, later reused as out staging
//   Y   124928 [64][80]  f32    (20480 B)
// ---------------------------------------------------------------------------
__global__ void __launch_bounds__(256)
attn_chunk_kernel(const __bf16* __restrict__ qp, const __bf16* __restrict__ kp,
                  const __bf16* __restrict__ v, const __bf16* __restrict__ states,
                  __bf16* __restrict__ attn) {
  extern __shared__ char smem[];
  __bf16* qpL = (__bf16*)smem;          // [64][256]
  __bf16* kpL = qpL + 64 * 256;         // [64][256]  (B^T for S=qp@kp^T)
  __bf16* PsL = kpL + 64 * 256;         // [80][256]  (B^T: state[dext][m])
  __bf16* vT  = PsL + 80 * 256;         // [80][64]
  __bf16* Sm  = vT + 80 * 64;           // [64][64]
  float*  Y   = (float*)(Sm + 64 * 64); // [64][80]
  const int t = (int)threadIdx.x, lane = t & 31, w = t >> 5;
  const int hl = lane >> 4, col = lane & 15;
  const int bid = (int)blockIdx.x;
  const int c = bid & (NCHK - 1), h = (bid >> 6) & (Hh - 1), b = bid >> 10;
  const int lbase = b * Ld + c * CHK;
  size_t gbase = (size_t)bid * SLOTS;
  const size_t crow = ((size_t)lbase * Hh + h) * Mm;  // qp/kp row for l=0

#if HAVE_TDM
  if (w == 0) {
    // 64 rows x 256 bf16, row stride H*M=4096 elems
    tdm_load_2d(0u,      qp + crow, 256u, 64u, 4096u);
    tdm_load_2d(32768u,  kp + crow, 256u, 64u, 4096u);
    // prefix state: contiguous 20480 bf16
    tdm_load_2d(65536u,  states + gbase, 20480u, 1u, 20480u);
    __builtin_amdgcn_s_wait_tensorcnt(0);
  }
#else
  for (int i = 0; i < 64; ++i) {
    int idx = i * 256 + t, l = idx >> 8, m = idx & 255;
    size_t ra = crow + (size_t)l * Hh * Mm + m;
    qpL[idx] = qp[ra];
    kpL[idx] = kp[ra];
  }
  for (int i = 0; i < 80; ++i) PsL[i * 256 + t] = states[gbase + i * 256 + t];
#endif
  for (int i = 0; i < 16; ++i) {
    int idx = i * 256 + t, l = idx >> 6, d = idx & 63;
    vT[d * 64 + l] = v[(size_t)(lbase + l) * HD + h * Dd + d];
  }
  for (int i = 0; i < 4; ++i) {
    int idx = i * 256 + t, rr = 64 + (idx >> 6);
    vT[rr * 64 + (idx & 63)] = (__bf16)((rr == 64) ? 1.0f : 0.0f);
  }
  __syncthreads();

  // Stage 1: inter-chunk Y = qp @ state   (20 tiles: 4x5, striped over 8 waves)
  v8f acc[3] = {};
  for (int kk = 0; kk < Mm; kk += 32) {
#pragma unroll
    for (int i = 0; i < 3; ++i) {
      int tt = w + 8 * i;
      if (tt >= 20) break;
      int tm = tt / 5, tn = tt % 5;
      v16bf a  = lda_frag(qpL, tm * 16, kk, 256);
      v16bf bb = ldb_frag(PsL, tn * 16, kk, 256);
      acc[i] = wmma_bf16(a, bb, acc[i]);
    }
  }

  // Stage 2: S = qp @ kp^T with causal mask (keep j <= i), 16 tiles (4x4)
  v8f sacc[2] = {};
  for (int kk = 0; kk < Mm; kk += 32) {
#pragma unroll
    for (int i = 0; i < 2; ++i) {
      int tt = w + 8 * i, tm = tt >> 2, tn = tt & 3;
      v16bf a  = lda_frag(qpL, tm * 16, kk, 256);
      v16bf bb = ldb_frag(kpL, tn * 16, kk, 256);
      sacc[i] = wmma_bf16(a, bb, sacc[i]);
    }
  }
#pragma unroll
  for (int i = 0; i < 2; ++i) {
    int tt = w + 8 * i, tm = tt >> 2, tn = tt & 3;
#pragma unroll
    for (int r = 0; r < 8; ++r) {
      int ri = tm * 16 + r + 8 * hl, cj = tn * 16 + col;
      float sv = (cj <= ri) ? sacc[i][r] : 0.f;
      Sm[ri * 64 + cj] = (__bf16)sv;
    }
  }
  __syncthreads();

  // Stage 3: Y += S @ v_ext
  for (int kk = 0; kk < CHK; kk += 32) {
#pragma unroll
    for (int i = 0; i < 3; ++i) {
      int tt = w + 8 * i;
      if (tt >= 20) break;
      int tm = tt / 5, tn = tt % 5;
      v16bf a  = lda_frag(Sm, tm * 16, kk, 64);
      v16bf bb = ldb_frag(vT, tn * 16, kk, 64);
      acc[i] = wmma_bf16(a, bb, acc[i]);
    }
  }

  // Stage 4: stage Y to LDS (need per-row denominator from column 64)
#pragma unroll
  for (int i = 0; i < 3; ++i) {
    int tt = w + 8 * i;
    if (tt >= 20) break;
    int tm = tt / 5, tn = tt % 5;
#pragma unroll
    for (int r = 0; r < 8; ++r)
      Y[(tm * 16 + r + 8 * hl) * DEXT + tn * 16 + col] = acc[i][r];
  }
  __syncthreads();

  // Stage 5: normalize; store attn[b,l,h,d]
#if HAVE_TDM
  // stage normalized bf16 tile into Sm (done being read after the barrier
  // above), then DMA it out: 64 rows x 64 bf16, row stride HD=1024 elems.
  for (int i = 0; i < 16; ++i) {
    int idx = i * 256 + t, l = idx >> 6, d = idx & 63;
    Sm[idx] = (__bf16)(Y[l * DEXT + d] / Y[l * DEXT + 64]);
  }
  __syncthreads();
  if (w == 0)
    tdm_store_2d(116736u, attn + (size_t)lbase * HD + h * Dd, 64u, 64u, 1024u);
  // implicit wait-idle at S_ENDPGM drains the issuing wave's TENSORcnt.
#else
  for (int i = 0; i < 16; ++i) {
    int idx = i * 256 + t, l = idx >> 6, d = idx & 63;
    attn[(size_t)(lbase + l) * HD + h * Dd + d] =
        (__bf16)(Y[l * DEXT + d] / Y[l * DEXT + 64]);
  }
#endif
}

// ---------------------------------------------------------------------------
extern "C" void kernel_launch(void* const* d_in, const int* in_sizes, int n_in,
                              void* d_out, int out_size, void* d_ws, size_t ws_size,
                              hipStream_t stream) {
  const float* x  = (const float*)d_in[0];
  const float* Wq = (const float*)d_in[1];
  const float* Wk = (const float*)d_in[2];
  const float* Wv = (const float*)d_in[3];
  const float* Wo = (const float*)d_in[4];
  const float* P  = (const float*)d_in[5];
  float* out = (float*)d_out;

  const size_t NQKV = (size_t)Bb * Ld * HD;           // 16,777,216
  const size_t NFEA = (size_t)Bb * Ld * Hh * Mm;      // 67,108,864

  __bf16* q      = (__bf16*)d_ws;
  __bf16* k      = q + NQKV;
  __bf16* v      = k + NQKV;
  __bf16* qp     = v + NQKV;
  __bf16* kp     = qp + NFEA;
  __bf16* attn   = kp + NFEA;
  __bf16* states = attn + NQKV;
  (void)ws_size; (void)in_sizes; (void)n_in; (void)out_size;

  dim3 blk(256);
  const int BL  = Bb * Ld;           // 16384 rows
  const int BLH = Bb * Ld * Hh;      // 262144 rows

  // K1: QKV projections (A fp32, B fp32 [K][N], out bf16)
  dim3 g1(HD / 64, BL / 128);
  gemm_kernel<float, __bf16, false, false><<<g1, blk, 0, stream>>>(x, Wq, q, BL, HD, HD, 1.f, 0.f);
  gemm_kernel<float, __bf16, false, false><<<g1, blk, 0, stream>>>(x, Wk, k, BL, HD, HD, 1.f, 0.f);
  gemm_kernel<float, __bf16, false, false><<<g1, blk, 0, stream>>>(x, Wv, v, BL, HD, HD, 1.f, 0.f);

  // K2: feature maps (A bf16, B = P already [N=M][K=D] transposed, relu epilogue)
  dim3 g2(Mm / 64, BLH / 128);
  const float ratio = 0.0625f;  // 1/sqrt(256)
  gemm_kernel<__bf16, __bf16, true, true><<<g2, blk, 0, stream>>>(q, P, qp, BLH, Mm, Dd, ratio, 1e-3f);
  gemm_kernel<__bf16, __bf16, true, true><<<g2, blk, 0, stream>>>(k, P, kp, BLH, Mm, Dd, ratio, 1e-3f);

  // K3a: local chunk states
  size_t lds3a = (256 * 64 + 80 * 64) * sizeof(__bf16);            // 43,008
  local_state_kernel<<<Bb * Hh * NCHK, blk, lds3a, stream>>>(kp, v, states);

  // K3b: exclusive prefix scan of states (in-place, transposed layout out)
  size_t lds3b = (size_t)SLOTS * sizeof(float);                    // 81,920
  scan_states_kernel<<<Bb * Hh, blk, lds3b, stream>>>(states);

  // K3c: chunked causal attention (TDM tile movement)
  size_t lds3c = (size_t)(64 * 256 + 64 * 256 + 80 * 256 + 80 * 64 + 64 * 64) * sizeof(__bf16)
               + (size_t)(64 * DEXT) * sizeof(float);              // 145,408
  attn_chunk_kernel<<<Bb * Hh * NCHK, blk, lds3c, stream>>>(qp, kp, v, states, attn);

  // K4: output projection -> fp32 d_out
  dim3 g4(HD / 64, BL / 128);
  gemm_kernel<__bf16, float, false, false><<<g4, blk, 0, stream>>>(attn, Wo, out, BL, HD, HD, 1.f, 0.f);
}
